// BiBNGRULayer_22943715295447
// MI455X (gfx1250) — compile-verified, matrix operand order checked by reading, not run on hardware
//
#include <hip/hip_runtime.h>
#include <hip/hip_bf16.h>
#include <math.h>

// ---------------------------------------------------------------------------
// Problem dims (fixed by the reference): T=1024, B=32, D=512, H=512, G=3H=1536
// M = T*B = 32768 rows for the projection GEMM.
// ---------------------------------------------------------------------------
#define T_DIM 1024
#define B_DIM 32
#define D_DIM 512
#define H_DIM 512
#define G_DIM 1536           // 3*H
#define M_DIM (T_DIM * B_DIM)

typedef __attribute__((ext_vector_type(16))) __bf16 v16bf;
typedef __attribute__((ext_vector_type(8)))  __bf16 v8bf;
typedef __attribute__((ext_vector_type(8)))  float  v8f;
typedef __attribute__((ext_vector_type(4)))  unsigned int v4u;
typedef __attribute__((ext_vector_type(8)))  unsigned int v8u;

// ---------------------------------------------------------------------------
// GRU-kernel LDS layout (per WGP, 320 KB available):
//   hbuf : f32  [32][520]   @ 0        (66,560 B)  stride%64==8 -> bank spread
//   hp   : bf16 [32][1560]  @ 66,560   (99,840 B)  dword stride 780 (%64==12)
//   xbuf : bf16 [32][1536]  @ 166,400  (98,304 B)  TDM destination tile
// ---------------------------------------------------------------------------
#define HSTR      520
#define HPSTR16   1560
#define HP_OFF    (32 * HSTR * 4)                       // 66,560
#define XBUF_OFF  (HP_OFF + 32 * HPSTR16 * 2)           // 166,400
#define GRU_LDS_BYTES (XBUF_OFF + B_DIM * G_DIM * 2)    // 264,704 < 320 KB

// ---------------------------------------------------------------------------
// Fragment helpers (CDNA5 WMMA 16x16x32 bf16, wave32).  ISA 7.12.2 layouts:
// A 16x32 (MxK): lane holds row M=lane&15; VGPR0-3 = K kb..kb+7,
// VGPR4-7 = K kb+16..kb+23, kb = (lane>=16)*8.
// B 32x16 (KxN): lane holds col N=lane&15 with the same K-half split; since
// B = W^T and W is row-major [G, K], a B column is contiguous in K.
// ---------------------------------------------------------------------------
static __device__ inline v16bf frag_from_bf16(const __bf16* __restrict__ p) {
  v8bf lo = *(const v8bf*)(p);
  v8bf hi = *(const v8bf*)(p + 16);
  v16bf r;
#pragma unroll
  for (int j = 0; j < 8; ++j) { r[j] = lo[j]; r[8 + j] = hi[j]; }
  return r;
}

static __device__ inline v16bf frag_from_f32(const float* __restrict__ p) {
  v16bf r;
#pragma unroll
  for (int j = 0; j < 8; ++j) r[j] = (__bf16)p[j];
#pragma unroll
  for (int j = 0; j < 8; ++j) r[8 + j] = (__bf16)p[16 + j];
  return r;
}

// ---------------------------------------------------------------------------
// Kernel 0: f32 -> bf16 weight conversion (one-time, tiny).
// ---------------------------------------------------------------------------
__global__ void cvt_f32_bf16_kernel(const float* __restrict__ in,
                                    __bf16* __restrict__ out, int n) {
  int i = blockIdx.x * blockDim.x + threadIdx.x;
  if (i < n) out[i] = (__bf16)in[i];
}

// ---------------------------------------------------------------------------
// Kernel 1: projection GEMM  xpb[M,G] = bf16( x[M,D] @ Wx^T )
// Block = 256 threads = 8 waves; block tile 128x64; wave tile 16x64.
// ---------------------------------------------------------------------------
__global__ void proj_gemm_kernel(const float* __restrict__ x,
                                 const __bf16* __restrict__ wxb,
                                 __bf16* __restrict__ xpb) {
  const int lane = threadIdx.x & 31;
  const int wv   = threadIdx.x >> 5;                 // 0..7
  const int row0 = blockIdx.x * 128 + wv * 16;
  const int col0 = blockIdx.y * 64;
  const int m    = lane & 15;
  const int kb   = (lane >> 4) * 8;

  v8f acc[4] = {};
  const float* arow = x + (size_t)(row0 + m) * D_DIM + kb;

  for (int k0 = 0; k0 < D_DIM; k0 += 32) {
    __builtin_prefetch(arow + k0 + 64, 0, 1);        // global_prefetch_b8
    v16bf a = frag_from_f32(arow + k0);
#pragma unroll
    for (int nt = 0; nt < 4; ++nt) {
      const __bf16* bp = wxb + (size_t)(col0 + nt * 16 + m) * D_DIM + k0 + kb;
      v16bf b = frag_from_bf16(bp);
      acc[nt] = __builtin_amdgcn_wmma_f32_16x16x32_bf16(
          false, a, false, b, (short)0, acc[nt], false, false);
    }
  }
#pragma unroll
  for (int nt = 0; nt < 4; ++nt) {
    const int col = col0 + nt * 16 + m;
#pragma unroll
    for (int r = 0; r < 8; ++r) {
      const int row = row0 + r + (lane >> 4) * 8;
      xpb[(size_t)row * G_DIM + col] = (__bf16)acc[nt][r];
    }
  }
}

// ---------------------------------------------------------------------------
// Kernel 2: BN stats folded into per-channel scale/shift.
// ---------------------------------------------------------------------------
__global__ void bn_stats_kernel(const __bf16* __restrict__ xpb,
                                const float* __restrict__ gamma,
                                const float* __restrict__ beta,
                                float* __restrict__ scale,
                                float* __restrict__ shift) {
  const int c = blockIdx.x * blockDim.x + threadIdx.x;
  if (c >= G_DIM) return;
  float s = 0.f, s2 = 0.f;
  for (int row = 0; row < M_DIM; ++row) {
    float v = (float)xpb[(size_t)row * G_DIM + c];
    s += v; s2 += v * v;
  }
  const float inv_n = 1.0f / (float)M_DIM;
  const float mean  = s * inv_n;
  const float var   = fmaxf(s2 * inv_n - mean * mean, 0.0f);
  const float sc    = gamma[c] * rsqrtf(var + 1e-5f);
  scale[c] = sc;
  shift[c] = beta[c] - mean * sc;
}

// ---------------------------------------------------------------------------
// Kernel 3: zero output (fwd/bwd accumulate via global_atomic_add_f32).
// ---------------------------------------------------------------------------
__global__ void zero_kernel(float* __restrict__ out, size_t n) {
  size_t i = (size_t)blockIdx.x * blockDim.x + threadIdx.x;
  size_t stride = (size_t)gridDim.x * blockDim.x;
  for (; i < n; i += stride) out[i] = 0.0f;
}

// ---------------------------------------------------------------------------
// TDM: async 1-D tile load (ISA ch.10 / D# groups, §8.3-8.4).
// Tile = 32*1536 bf16 = 49152 elems (contiguous), data_size code 1 (2 bytes).
// Group0: dword0 count=1 | dword1 lds_addr | dword2/3 global_addr, type=2.
// Group1: data_size<<16 | tensor_dim0 @bits79:48 | tensor_dim1=1 | tile_dim0.
// Groups 2/3 are NULL (<=2-D tensor).  Tracked with TENSORcnt.
// ---------------------------------------------------------------------------
static __device__ inline void tdm_load_tile(const __bf16* gsrc, unsigned lds_off) {
  const unsigned long long ga = (unsigned long long)(size_t)gsrc;
  const unsigned n_elem = B_DIM * G_DIM;               // 49152 < 2^16
  v4u g0;
  g0[0] = 1u;                                          // count=1, user D#
  g0[1] = lds_off;                                     // LDS byte address
  g0[2] = (unsigned)ga;                                // global_addr[31:0]
  g0[3] = (unsigned)((ga >> 32) & 0x1FFFFFFu) | (2u << 30);  // addr[56:32]|type=2
  v8u g1;
  g1[0] = 1u << 16;                                    // data_size=1 (2B)
  g1[1] = (n_elem & 0xFFFFu) << 16;                    // tensor_dim0[15:0]
  g1[2] = ((n_elem >> 16) & 0xFFFFu) | (1u << 16);     // dim0[31:16] | tensor_dim1=1
  g1[3] = (n_elem & 0xFFFFu) << 16;                    // tile_dim0
  g1[4] = 0u;                                          // tile_dim1/2 unused
  g1[5] = n_elem;                                      // tensor_dim0_stride[31:0]
  g1[6] = 0u;
  g1[7] = 0u;
  asm volatile("tensor_load_to_lds %0, %1"
               :
               : "s"(g0), "s"(g1)
               : "memory");
}

// ---------------------------------------------------------------------------
// Kernel 4: persistent bidirectional GRU scan.
// grid.x = 2 (0 = fwd, 1 = bwd), block = 1024 threads = 32 wave32s = 1 WGP.
// Per step: TDM prefetch of x_proj[t] into LDS overlaps the WMMA GEMM
// hp[32,1536] = h[32,512] @ Wh^T (3072 v_wmma per step), then gates.
// ---------------------------------------------------------------------------
__global__ void gru_scan_kernel(const __bf16* __restrict__ xpb,
                                const __bf16* __restrict__ whfb,
                                const __bf16* __restrict__ whbb,
                                const float* __restrict__ scale,
                                const float* __restrict__ shift,
                                float* __restrict__ out) {
  extern __shared__ float smem[];
  float*  hbuf = smem;                                   // [32][HSTR] f32
  __bf16* hp   = (__bf16*)((char*)smem + HP_OFF);        // [32][HPSTR16] bf16
  __bf16* xbuf = (__bf16*)((char*)smem + XBUF_OFF);      // [32][1536] bf16

  const int dir  = blockIdx.x;
  const __bf16* wh = dir ? whbb : whfb;
  const int tid  = threadIdx.x;
  const int lane = tid & 31;
  const int wv   = tid >> 5;           // 0..31
  const int mtile = wv & 1;
  const int ncol0 = (wv >> 1) * 96;    // 16 groups x 96 cols = 1536
  const int m    = lane & 15;
  const int kb   = (lane >> 4) * 8;

  for (int i = tid; i < 32 * HSTR; i += 1024) hbuf[i] = 0.0f;
  __syncthreads();

  const float* arow = hbuf + (size_t)(mtile * 16 + m) * HSTR + kb;

  for (int step = 0; step < T_DIM; ++step) {
    const int t = dir ? (T_DIM - 1 - step) : step;

    // ---- issue async x_proj[t] tile fetch (overlaps the GEMM below) ----
    if (wv == 0)
      tdm_load_tile(xpb + (size_t)t * B_DIM * G_DIM, XBUF_OFF);

    // ---- hp = h @ Wh^T (this wave's 1x6 tile strip) ----
    v8f acc[6] = {};
    for (int k0 = 0; k0 < H_DIM; k0 += 32) {
      v16bf a = frag_from_f32(arow + k0);            // LDS f32 -> bf16 frag
#pragma unroll
      for (int nt = 0; nt < 6; ++nt) {
        const __bf16* bp = wh + (size_t)(ncol0 + nt * 16 + m) * H_DIM + k0 + kb;
        v16bf b = frag_from_bf16(bp);
        acc[nt] = __builtin_amdgcn_wmma_f32_16x16x32_bf16(
            false, a, false, b, (short)0, acc[nt], false, false);
      }
    }
#pragma unroll
    for (int nt = 0; nt < 6; ++nt) {
      const int col = ncol0 + nt * 16 + m;
#pragma unroll
      for (int r = 0; r < 8; ++r) {
        const int row = mtile * 16 + r + (lane >> 4) * 8;
        hp[row * HPSTR16 + col] = (__bf16)acc[nt][r];
      }
    }
    if (wv == 0)
      __builtin_amdgcn_s_wait_tensorcnt(0);          // x tile landed in LDS
    __syncthreads();

    // ---- gates: r,z,n and state update; 16384 elems / 1024 threads ----
    for (int e = tid; e < B_DIM * H_DIM; e += 1024) {
      const int b = e >> 9;            // batch
      const int j = e & 511;           // hidden idx
      const __bf16* xb = xbuf + b * G_DIM;
      const float xr = (float)xb[j]        * scale[j]        + shift[j];
      const float xz = (float)xb[j + 512]  * scale[j + 512]  + shift[j + 512];
      const float xn = (float)xb[j + 1024] * scale[j + 1024] + shift[j + 1024];
      const float hr = (float)hp[b * HPSTR16 + j];
      const float hz = (float)hp[b * HPSTR16 + j + 512];
      const float hn = (float)hp[b * HPSTR16 + j + 1024];
      const float rg = 1.0f / (1.0f + __expf(-(xr + hr)));
      const float zg = 1.0f / (1.0f + __expf(-(xz + hz)));
      const float ng = tanhf(xn + rg * hn);
      const float hprev = hbuf[b * HSTR + j];
      const float hnew  = (1.0f - zg) * hprev + zg * ng;
      hbuf[b * HSTR + j] = hnew;
      atomicAdd(out + (size_t)t * (B_DIM * H_DIM) + e, hnew);
    }
    __syncthreads();   // gate reads of xbuf/hbuf done before next TDM/GEMM
  }
}

// ---------------------------------------------------------------------------
// Host-side launch.  Inputs: x, Wx, Wh_fwd, Wh_bwd, gamma, beta (all f32).
// Workspace: xpb bf16 [M,G] | wxb/whfb/whbb bf16 | scale/shift f32.
// ---------------------------------------------------------------------------
extern "C" void kernel_launch(void* const* d_in, const int* in_sizes, int n_in,
                              void* d_out, int out_size, void* d_ws, size_t ws_size,
                              hipStream_t stream) {
  const float* x     = (const float*)d_in[0];
  const float* Wx    = (const float*)d_in[1];
  const float* Whf   = (const float*)d_in[2];
  const float* Whb   = (const float*)d_in[3];
  const float* gamma = (const float*)d_in[4];
  const float* beta  = (const float*)d_in[5];
  float* out = (float*)d_out;

  char* ws = (char*)d_ws;
  size_t off = 0;
  __bf16* xpb  = (__bf16*)(ws + off); off += (size_t)M_DIM * G_DIM * 2;
  __bf16* wxb  = (__bf16*)(ws + off); off += (size_t)G_DIM * D_DIM * 2;
  __bf16* whfb = (__bf16*)(ws + off); off += (size_t)G_DIM * H_DIM * 2;
  __bf16* whbb = (__bf16*)(ws + off); off += (size_t)G_DIM * H_DIM * 2;
  float*  scale = (float*)(ws + off); off += G_DIM * 4;
  float*  shift = (float*)(ws + off); off += G_DIM * 4;
  (void)ws_size; (void)in_sizes; (void)n_in;

  const int wn = G_DIM * D_DIM;                 // 786432 (D == H)
  cvt_f32_bf16_kernel<<<(wn + 255) / 256, 256, 0, stream>>>(Wx,  wxb,  wn);
  cvt_f32_bf16_kernel<<<(wn + 255) / 256, 256, 0, stream>>>(Whf, whfb, wn);
  cvt_f32_bf16_kernel<<<(wn + 255) / 256, 256, 0, stream>>>(Whb, whbb, wn);

  dim3 pg(M_DIM / 128, G_DIM / 64);             // (256, 24)
  proj_gemm_kernel<<<pg, 256, 0, stream>>>(x, wxb, xpb);

  bn_stats_kernel<<<G_DIM / 256, 256, 0, stream>>>(xpb, gamma, beta, scale, shift);

  zero_kernel<<<1024, 256, 0, stream>>>(out, (size_t)out_size);

  gru_scan_kernel<<<2, 1024, GRU_LDS_BYTES, stream>>>(xpb, whfb, whbb,
                                                      scale, shift, out);
}
